// AttentionDec_33182917328988
// MI455X (gfx1250) — matrix operand order repaired and checked by reading, not compiled, not run on hardware
//
#include <hip/hip_runtime.h>
#include <hip/hip_bf16.h>

// ---------------------------------------------------------------------------
// CDNA5 (gfx1250) WMMA attention-decoder step.
// All GEMMs run through v_wmma_f32_16x16x32_bf16 (f32->bf16 convert in regs).
// Wave tile = 32x32 (2 A-frags x 2 B-frags -> 4 WMMAs). Explicit ping-pong
// software pipeline (unroll-by-2, two static buffer sets) so next-chunk loads
// fly behind the WMMAs without loop-carried register copies.
// ---------------------------------------------------------------------------

typedef __attribute__((ext_vector_type(16))) __bf16 v16bf;
typedef __attribute__((ext_vector_type(8)))  float  v8f;
typedef __attribute__((ext_vector_type(4)))  float  v4f;

#define NT 256
#define NB 256
#define HH 1024
#define VV 32000

static __device__ __forceinline__ v4f ld4(const float* __restrict__ p,
                                          unsigned off) {
  return *(const v4f*)(p + off);
}

static __device__ __forceinline__ void cvt16(v16bf& r, const v4f f[4]) {
#pragma unroll
  for (int q = 0; q < 4; q++)
#pragma unroll
    for (int i = 0; i < 4; i++) r[q * 4 + i] = (__bf16)f[q][i];
}

static __device__ __forceinline__ v8f wmma_bf16(v16bf a, v16bf b, v8f c) {
  return __builtin_amdgcn_wmma_f32_16x16x32_bf16(false, a, false, b,
                                                 (short)0, c, false, false);
}

static __device__ __forceinline__ float sigmoidf(float x) {
  return 1.0f / (1.0f + expf(-x));
}

// A fragment raw offsets (16x32 bf16, ISA 7.12.2): lane = row (lane&15);
// lanes 0-15 hold K {k0..k0+7, k0+16..k0+23}, lanes 16-31 K {+8..15, +24..31}.
static __device__ __forceinline__ void load_a_raw(v4f (&ra)[4],
                                                  const float* __restrict__ A,
                                                  unsigned o) {
  ra[0] = ld4(A, o);
  ra[1] = ld4(A, o + 4);
  ra[2] = ld4(A, o + 16);
  ra[3] = ld4(A, o + 20);
}

// B fragment raw offsets (32x16 bf16 from row-major W[N][K]): lane = column;
// lanes 0-15 hold K {k0..k0+7, k0+16..k0+23}, lanes 16-31 K {+8..15, +24..31}.
static __device__ __forceinline__ void load_b_raw(v4f (&rb)[4],
                                                  const float* __restrict__ B,
                                                  unsigned o) {
  rb[0] = ld4(B, o);
  rb[1] = ld4(B, o + 4);
  rb[2] = ld4(B, o + 16);
  rb[3] = ld4(B, o + 20);
}

// One pipeline phase: (optionally) issue next chunk's 4 fragment loads into
// `nxt*`, then convert `cur*` (loaded last phase) and run the 4 WMMAs.
template <int K>
static __device__ __forceinline__ void pipe_phase(
    const float* __restrict__ A, const float* __restrict__ B, unsigned a0,
    unsigned b0, v4f (&cA0)[4], v4f (&cA1)[4], v4f (&cB0)[4], v4f (&cB1)[4],
    v4f (&nA0)[4], v4f (&nA1)[4], v4f (&nB0)[4], v4f (&nB1)[4], int knext,
    bool do_load, v8f acc[4]) {
  if (do_load) {
    load_a_raw(nA0, A, a0 + (unsigned)knext);
    load_a_raw(nA1, A, a0 + (unsigned)(16 * K + knext));
    load_b_raw(nB0, B, b0 + (unsigned)knext);
    load_b_raw(nB1, B, b0 + (unsigned)(16 * K + knext));
  }
  v16bf a0f, a1f, b0f, b1f;
  cvt16(a0f, cA0);
  cvt16(a1f, cA1);
  cvt16(b0f, cB0);
  cvt16(b1f, cB1);
  acc[0] = wmma_bf16(a0f, b0f, acc[0]);
  acc[1] = wmma_bf16(a0f, b1f, acc[1]);
  acc[2] = wmma_bf16(a1f, b0f, acc[2]);
  acc[3] = wmma_bf16(a1f, b1f, acc[3]);
}

// ---------------------------------------------------------------------------
// Pipelined 32x32 tile accumulation over K (compile-time, K % 64 == 0).
// a0 = (m0 + lane%16)*K + 8*(lane/16)   (second m-half at a0 + 16*K)
// b0 = (n0 + lane%16)*K + 8*(lane/16)   (second n-half at b0 + 16*K)
// acc[0]=m0,n0  acc[1]=m0,n16  acc[2]=m16,n0  acc[3]=m16,n16
// ---------------------------------------------------------------------------
template <int K>
static __device__ __forceinline__ void wmma_tile_32x32(
    const float* __restrict__ A, unsigned a0, const float* __restrict__ B,
    unsigned b0, v8f acc[4]) {
  v4f pA0[4], pA1[4], pB0[4], pB1[4];  // ping
  v4f qA0[4], qA1[4], qB0[4], qB1[4];  // pong
  load_a_raw(pA0, A, a0);
  load_a_raw(pA1, A, a0 + (unsigned)(16 * K));
  load_b_raw(pB0, B, b0);
  load_b_raw(pB1, B, b0 + (unsigned)(16 * K));

#pragma unroll 1
  for (int k0 = 0; k0 < K; k0 += 64) {
    // consume ping, prefetch k0+32 into pong (always in range: k0 <= K-64)
    pipe_phase<K>(A, B, a0, b0, pA0, pA1, pB0, pB1, qA0, qA1, qB0, qB1,
                  k0 + 32, true, acc);
    // consume pong, prefetch k0+64 into ping
    pipe_phase<K>(A, B, a0, b0, qA0, qA1, qB0, qB1, pA0, pA1, pB0, pB1,
                  k0 + 64, (k0 + 64) < K, acc);
  }
}

// ---------------------------------------------------------------------------
// Generic C[M,N] = A[M,K] @ B[N,K]^T + bias[N].  One wave owns a 32x32 tile.
// M%32==0, N%32==0, K%64==0. blockDim.x = 128 (4 waves).
// ---------------------------------------------------------------------------
template <int K>
__global__ __launch_bounds__(128, 1) void gemm_abt_wmma(
    const float* __restrict__ A, const float* __restrict__ B,
    const float* __restrict__ bias, float* __restrict__ C, int M, int N) {
  const int lane = threadIdx.x & 31;
  const int wave = threadIdx.x >> 5;
  const int tile = blockIdx.x * (blockDim.x >> 5) + wave;
  const int mtiles = M >> 5;
  const int ntiles = N >> 5;
  if (tile >= mtiles * ntiles) return;
  const int m0 = (tile % mtiles) << 5;
  const int n0 = (tile / mtiles) << 5;

  const int hi = lane >> 4;
  const int nc = lane & 15;
  const unsigned a0 = (unsigned)(m0 + nc) * K + 8u * hi;
  const unsigned b0 = (unsigned)(n0 + nc) * K + 8u * hi;

  v8f acc[4] = {};
  wmma_tile_32x32<K>(A, a0, B, b0, acc);

  // C/D layout: lane l, vgpr v -> row (+v + 8*(l>>4)), col (+ (l&15))
#pragma unroll
  for (int v = 0; v < 8; v++) {
    float* crow0 = C + (size_t)(m0 + v + 8 * hi) * N;
    float* crow1 = C + (size_t)(m0 + 16 + v + 8 * hi) * N;
    const float bia0 = bias[n0 + nc];
    const float bia1 = bias[n0 + 16 + nc];
    crow0[n0 + nc]      = acc[0][v] + bia0;
    crow0[n0 + 16 + nc] = acc[1][v] + bia1;
    crow1[n0 + nc]      = acc[2][v] + bia0;
    crow1[n0 + 16 + nc] = acc[3][v] + bia1;
  }
}

// ---------------------------------------------------------------------------
// Fused attention-energy kernel:
//   e[t*NB+b] = sum_o tanh( (enc[t,b,:]@W_i2h[o,:]) + hproj[b,o] ) * wscore[o]
// 137 GFLOP GEMM, fused reduction (never materializes the 268MB enc_proj).
// One wave = 32 rows (one t, 32 consecutive b).  blockDim.x = 128 (4 waves).
// ---------------------------------------------------------------------------
__global__ __launch_bounds__(128, 1) void energy_wmma(
    const float* __restrict__ enc, const float* __restrict__ Wi2h,
    const float* __restrict__ hproj, const float* __restrict__ wscore,
    float* __restrict__ e) {
  const int lane = threadIdx.x & 31;
  const int wave = threadIdx.x >> 5;
  const int tile = blockIdx.x * (blockDim.x >> 5) + wave;  // 0..2047
  const int r0 = tile << 5;  // flat row = t*NB + b, 32 rows, same t
  const int b0 = r0 & (NB - 1);
  const int hi = lane >> 4;
  const int nc = lane & 15;
  const unsigned a0 = (unsigned)(r0 + nc) * HH + 8u * hi;

  float eacc0[8] = {0.f, 0.f, 0.f, 0.f, 0.f, 0.f, 0.f, 0.f};  // rows b0+..
  float eacc1[8] = {0.f, 0.f, 0.f, 0.f, 0.f, 0.f, 0.f, 0.f};  // rows b0+16+..

#pragma unroll 1
  for (int n0 = 0; n0 < HH; n0 += 32) {
    const unsigned bo = (unsigned)(n0 + nc) * HH + 8u * hi;
    v8f acc[4] = {};
    wmma_tile_32x32<HH>(enc, a0, Wi2h, bo, acc);

    // fused epilogue: tanh(+hproj) * wscore, accumulate per-row
    const float wsc0 = wscore[n0 + nc];
    const float wsc1 = wscore[n0 + 16 + nc];
    const float* hp0 = hproj + (size_t)(b0 + 8 * hi) * HH + n0 + nc;
    const float* hp1 = hproj + (size_t)(b0 + 16 + 8 * hi) * HH + n0 + nc;
#pragma unroll
    for (int v = 0; v < 8; v++) {
      eacc0[v] += tanhf(acc[0][v] + hp0[(size_t)v * HH]) * wsc0;
      eacc0[v] += tanhf(acc[1][v] + hp0[(size_t)v * HH + 16]) * wsc1;
      eacc1[v] += tanhf(acc[2][v] + hp1[(size_t)v * HH]) * wsc0;
      eacc1[v] += tanhf(acc[3][v] + hp1[(size_t)v * HH + 16]) * wsc1;
    }
  }

  // reduce the 16 column-lanes of each half-wave, write 32 energies
#pragma unroll
  for (int v = 0; v < 8; v++) {
    float s0 = eacc0[v];
    s0 += __shfl_xor(s0, 1, 32);
    s0 += __shfl_xor(s0, 2, 32);
    s0 += __shfl_xor(s0, 4, 32);
    s0 += __shfl_xor(s0, 8, 32);
    float s1 = eacc1[v];
    s1 += __shfl_xor(s1, 1, 32);
    s1 += __shfl_xor(s1, 2, 32);
    s1 += __shfl_xor(s1, 4, 32);
    s1 += __shfl_xor(s1, 8, 32);
    if (nc == 0) {
      e[r0 + v + 8 * hi] = s0;
      e[r0 + 16 + v + 8 * hi] = s1;
    }
  }
}

// ---------------------------------------------------------------------------
// alpha[b,t] = softmax_t( e[t*NB+b] ).   grid = NB blocks, 256 threads.
// ---------------------------------------------------------------------------
__global__ void softmax_t_kernel(const float* __restrict__ e,
                                 float* __restrict__ alpha) {
  const int b = blockIdx.x;
  const int t = threadIdx.x;
  __shared__ float red[NT];
  const float v = e[(size_t)t * NB + b];
  red[t] = v;
  __syncthreads();
  for (int s = NT / 2; s > 0; s >>= 1) {
    if (t < s) red[t] = fmaxf(red[t], red[t + s]);
    __syncthreads();
  }
  const float m = red[0];
  __syncthreads();
  const float ex = expf(v - m);
  red[t] = ex;
  __syncthreads();
  for (int s = NT / 2; s > 0; s >>= 1) {
    if (t < s) red[t] += red[t + s];
    __syncthreads();
  }
  alpha[(size_t)b * NT + t] = ex / red[0];
}

// ---------------------------------------------------------------------------
// x[b, 1024+i] = emb[tokens[b], i]   (embedding gather half of the concat)
// ---------------------------------------------------------------------------
__global__ void gather_emb_kernel(const int* __restrict__ tokens,
                                  const float* __restrict__ emb,
                                  float* __restrict__ x) {
  const int b = blockIdx.x;
  const size_t tok = (size_t)tokens[b];
  for (int i = threadIdx.x; i < HH; i += blockDim.x)
    x[(size_t)b * (2 * HH) + HH + i] = emb[tok * HH + i];
}

// ---------------------------------------------------------------------------
// x[b, h] = sum_t alpha[b,t] * enc[t,b,h]   grid = (NB, HH/256), 256 threads.
// ---------------------------------------------------------------------------
__global__ void context_kernel(const float* __restrict__ alpha,
                               const float* __restrict__ enc,
                               float* __restrict__ x) {
  const int b = blockIdx.x;
  const int h = blockIdx.y * blockDim.x + threadIdx.x;
  __shared__ float al[NT];
  al[threadIdx.x] = alpha[(size_t)b * NT + threadIdx.x];
  __syncthreads();
  float acc = 0.f;
  for (int t = 0; t < NT; t++)
    acc += al[t] * enc[((size_t)t * NB + b) * HH + h];
  x[(size_t)b * (2 * HH) + h] = acc;
}

// ---------------------------------------------------------------------------
// GRU gate fusion (PyTorch order r,z,n):  h_new -> out region 2.
// ---------------------------------------------------------------------------
__global__ void gru_kernel(const float* __restrict__ gi,
                           const float* __restrict__ gh,
                           const float* __restrict__ h0,
                           float* __restrict__ hnew) {
  const int b = blockIdx.x;
  for (int i = threadIdx.x; i < HH; i += blockDim.x) {
    const size_t g = (size_t)b * (3 * HH) + i;
    const float r = sigmoidf(gi[g] + gh[g]);
    const float z = sigmoidf(gi[g + HH] + gh[g + HH]);
    const float n = tanhf(gi[g + 2 * HH] + r * gh[g + 2 * HH]);
    const float h = h0[(size_t)b * HH + i];
    hnew[(size_t)b * HH + i] = (1.0f - z) * n + z * h;
  }
}

// ---------------------------------------------------------------------------
// out[b,:] = log_softmax(logits[b,:])   grid = NB blocks, 256 threads.
// ---------------------------------------------------------------------------
__global__ void logsoftmax_kernel(const float* __restrict__ logits,
                                  float* __restrict__ out) {
  const int b = blockIdx.x;
  const int tid = threadIdx.x;
  __shared__ float red[256];
  const float* row = logits + (size_t)b * VV;
  float m = -3.402823466e38f;
  for (int v = tid; v < VV; v += 256) m = fmaxf(m, row[v]);
  red[tid] = m;
  __syncthreads();
  for (int s = 128; s > 0; s >>= 1) {
    if (tid < s) red[tid] = fmaxf(red[tid], red[tid + s]);
    __syncthreads();
  }
  m = red[0];
  __syncthreads();
  float sum = 0.f;
  for (int v = tid; v < VV; v += 256) sum += expf(row[v] - m);
  red[tid] = sum;
  __syncthreads();
  for (int s = 128; s > 0; s >>= 1) {
    if (tid < s) red[tid] += red[tid + s];
    __syncthreads();
  }
  const float lse = logf(red[0]) + m;
  float* orow = out + (size_t)b * VV;
  for (int v = tid; v < VV; v += 256) orow[v] = row[v] - lse;
}

// ---------------------------------------------------------------------------
extern "C" void kernel_launch(void* const* d_in, const int* in_sizes, int n_in,
                              void* d_out, int out_size, void* d_ws,
                              size_t ws_size, hipStream_t stream) {
  (void)in_sizes; (void)n_in; (void)out_size; (void)ws_size;

  const int*   tokens  = (const int*)d_in[0];
  const float* h0      = (const float*)d_in[1];   // [1,NB,H] -> [NB,H]
  const float* enc     = (const float*)d_in[2];   // [NT,NB,H]
  const float* emb     = (const float*)d_in[3];   // [V,H]
  const float* W_i2h   = (const float*)d_in[4];   // [H,H]
  const float* W_h2h   = (const float*)d_in[5];   // [H,H]
  const float* b_h2h   = (const float*)d_in[6];   // [H]
  const float* w_score = (const float*)d_in[7];   // [H]
  const float* W_ih    = (const float*)d_in[8];   // [3H,2H]
  const float* b_ih    = (const float*)d_in[9];   // [3H]
  const float* W_hh    = (const float*)d_in[10];  // [3H,H]
  const float* b_hh    = (const float*)d_in[11];  // [3H]
  const float* W_out   = (const float*)d_in[12];  // [V,H]
  const float* b_out   = (const float*)d_in[13];  // [V]

  float* out_lsm   = (float*)d_out;                       // [NB,V]
  float* out_h     = out_lsm + (size_t)NB * VV;           // [1,NB,H]
  float* out_alpha = out_h + (size_t)NB * HH;             // [NB,NT]

  float* ws     = (float*)d_ws;
  float* hproj  = ws;                                     // NB*H
  float* e      = hproj + (size_t)NB * HH;                // NT*NB
  float* x      = e + (size_t)NT * NB;                    // NB*2H
  float* gi     = x + (size_t)NB * 2 * HH;                // NB*3H
  float* gh     = gi + (size_t)NB * 3 * HH;               // NB*3H
  float* logits = gh + (size_t)NB * 3 * HH;               // NB*V

  // 1. hproj = h0 @ W_h2h^T + b_h2h              [NB,H]
  {
    int tiles = (NB / 32) * (HH / 32);
    gemm_abt_wmma<HH><<<(tiles + 3) / 4, 128, 0, stream>>>(h0, W_h2h, b_h2h,
                                                           hproj, NB, HH);
  }
  // 2. fused energy GEMM -> e[t*NB+b]
  {
    int tiles = (NT * NB) / 32;  // 2048 wave-tiles
    energy_wmma<<<tiles / 4, 128, 0, stream>>>(enc, W_i2h, hproj, w_score, e);
  }
  // 3. alpha = softmax_t(e^T)                    -> out region 3
  softmax_t_kernel<<<NB, NT, 0, stream>>>(e, out_alpha);
  // 4. x[:,H:] = emb[tokens]
  gather_emb_kernel<<<NB, 256, 0, stream>>>(tokens, emb, x);
  // 5. x[:,:H] = context = alpha @ enc
  context_kernel<<<dim3(NB, HH / 256), 256, 0, stream>>>(out_alpha, enc, x);
  // 6. gi = x @ W_ih^T + b_ih                    [NB,3H], K=2H
  {
    int tiles = (NB / 32) * ((3 * HH) / 32);
    gemm_abt_wmma<2 * HH><<<(tiles + 3) / 4, 128, 0, stream>>>(x, W_ih, b_ih,
                                                               gi, NB, 3 * HH);
  }
  // 7. gh = h0 @ W_hh^T + b_hh                   [NB,3H], K=H
  {
    int tiles = (NB / 32) * ((3 * HH) / 32);
    gemm_abt_wmma<HH><<<(tiles + 3) / 4, 128, 0, stream>>>(h0, W_hh, b_hh, gh,
                                                           NB, 3 * HH);
  }
  // 8. GRU gates -> h_new (out region 2)
  gru_kernel<<<NB, 256, 0, stream>>>(gi, gh, h0, out_h);
  // 9. logits = h_new @ W_out^T + b_out          [NB,V]
  {
    int tiles = (NB / 32) * (VV / 32);  // 8000
    gemm_abt_wmma<HH><<<(tiles + 3) / 4, 128, 0, stream>>>(out_h, W_out, b_out,
                                                           logits, NB, VV);
  }
  // 10. out = log_softmax(logits)
  logsoftmax_kernel<<<NB, 256, 0, stream>>>(logits, out_lsm);
}
